// Model4_39067022524749
// MI455X (gfx1250) — compile-verified, matrix operand order checked by reading, not run on hardware
//
#include <hip/hip_runtime.h>

// Problem constants (from reference): V=32000, E=512, H=512, C=8, B=64, T=512
#define E_DIM 512
#define H_DIM 512
#define B_DIM 64
#define T_DIM 512
#define C_DIM 8
#define NWG_REC 8        // persistent workgroups in the scan kernel
#define KP 520           // padded K stride (elements) for LDS tiles in scan
#define P1_LDK 36        // padded K stride for phase-1 LDS tiles

typedef __attribute__((ext_vector_type(8)))  float  v8f;
typedef __attribute__((ext_vector_type(16))) __bf16 v16bf;
typedef __attribute__((ext_vector_type(4)))  unsigned v4u;
typedef __attribute__((ext_vector_type(8)))  int      v8i_t;
typedef __attribute__((ext_vector_type(4)))  int      v4i_t;

#if __has_builtin(__builtin_amdgcn_tensor_load_to_lds) && \
    __has_builtin(__builtin_amdgcn_s_wait_tensorcnt)
#define USE_TDM 1
#else
#define USE_TDM 0
#endif

__device__ __forceinline__ v8f vzero8() {
  v8f v;
#pragma unroll
  for (int i = 0; i < 8; ++i) v[i] = 0.0f;
  return v;
}
__device__ __forceinline__ v8f vsplat8(float x) {
  v8f v;
#pragma unroll
  for (int i = 0; i < 8; ++i) v[i] = x;
  return v;
}

// Load a 16x32 bf16 fragment (A layout, ISA 7.12.2) from an LDS tile.
// lanes 0..15:  row=lane,    K = {0..7} U {16..23}
// lanes 16..31: row=lane-16, K = {8..15} U {24..31}
// B fragments use the same loader on a tile stored transposed ([N][K]).
__device__ __forceinline__ v16bf frag_ld(const __bf16* base, int ld, int lane) {
  const int r  = lane & 15;
  const int kb = (lane & 16) ? 8 : 0;
  const __bf16* p = base + r * ld + kb;
  v16bf f;
#pragma unroll
  for (int i = 0; i < 8; ++i) f[i] = p[i];          // K = kb + i
#pragma unroll
  for (int i = 0; i < 8; ++i) f[8 + i] = p[16 + i]; // K = kb + 16 + i
  return f;
}

__device__ __forceinline__ v8f wmma_bf16(v16bf a, v16bf b, v8f c) {
  // (neg_a, A, neg_b, B, c_mod, C, reuse_a, reuse_b)
  return __builtin_amdgcn_wmma_f32_16x16x32_bf16(false, a, false, b, (short)0, c,
                                                 false, false);
}

// ---------------------------------------------------------------------------
// Phase 1: xin[t][b][h] = (emb[x[b,t]] @ W1x)[h] + b1[h]
// 32768 x 512 x 512 GEMM with fused embedding gather, bf16 WMMA / f32 acc.
// Grid: (ROWS/64, H/64), 128 threads (4 waves); each wave = 16x64 strip.
// ---------------------------------------------------------------------------
__global__ __launch_bounds__(128) void phase1_gemm(
    const int* __restrict__ x, const float* __restrict__ emb,
    const float* __restrict__ W1, const float* __restrict__ b1,
    float* __restrict__ xin) {
  __shared__ int    sTok[64];
  __shared__ __bf16 sA[64 * P1_LDK];  // gathered embedding tile [row][k]
  __shared__ __bf16 sB[64 * P1_LDK];  // W1x tile, transposed     [n][k]

  const int tid  = threadIdx.x;
  const int lane = tid & 31;
  const int wave = tid >> 5;
  const int row0 = blockIdx.x * 64;   // flat (b*T + t) row base
  const int col0 = blockIdx.y * 64;   // H column base

  if (tid < 64) sTok[tid] = x[row0 + tid];

  v8f acc[4];
#pragma unroll
  for (int i = 0; i < 4; ++i) acc[i] = vzero8();

  for (int k0 = 0; k0 < E_DIM; k0 += 32) {
    __syncthreads();  // also covers sTok on first iteration
    // A tile: 64 rows x 32 k, gathered as float4 (8 per row)
#pragma unroll 4
    for (int e = tid; e < 64 * 8; e += 128) {
      const int i = e >> 3, kq = (e & 7) * 4;
      const float4 v = *(const float4*)&emb[(size_t)sTok[i] * E_DIM + k0 + kq];
      __bf16* d = &sA[i * P1_LDK + kq];
      d[0] = (__bf16)v.x; d[1] = (__bf16)v.y;
      d[2] = (__bf16)v.z; d[3] = (__bf16)v.w;
    }
    // B tile: 32 k-rows x 64 n as float4, stored transposed [n][k]
#pragma unroll 4
    for (int e = tid; e < 32 * 16; e += 128) {
      const int k = e >> 4, nq = (e & 15) * 4;
      const float4 v = *(const float4*)&W1[(size_t)(k0 + k) * H_DIM + col0 + nq];
      sB[(nq + 0) * P1_LDK + k] = (__bf16)v.x;
      sB[(nq + 1) * P1_LDK + k] = (__bf16)v.y;
      sB[(nq + 2) * P1_LDK + k] = (__bf16)v.z;
      sB[(nq + 3) * P1_LDK + k] = (__bf16)v.w;
    }
    if (k0 + 32 < E_DIM)
      __builtin_prefetch(&W1[(k0 + 32) * H_DIM + col0], 0, 1);  // global_prefetch
    __syncthreads();

    const v16bf a = frag_ld(&sA[(wave * 16) * P1_LDK], P1_LDK, lane);
#pragma unroll
    for (int nc = 0; nc < 4; ++nc) {
      const v16bf b = frag_ld(&sB[(nc * 16) * P1_LDK], P1_LDK, lane);
      acc[nc] = wmma_bf16(a, b, acc[nc]);
    }
  }

  // Store to (T, B, H) layout so each scan step reads a contiguous 64x512 slab
  const int mrow = (lane & 16) ? 8 : 0;
#pragma unroll
  for (int nc = 0; nc < 4; ++nc) {
    const int col = col0 + nc * 16 + (lane & 15);
    const float bb = b1[col];
#pragma unroll
    for (int r = 0; r < 8; ++r) {
      const int grow  = row0 + wave * 16 + mrow + r;
      const int b_idx = grow >> 9;   // / T
      const int t_idx = grow & 511;  // % T
      xin[((size_t)t_idx * B_DIM + b_idx) * H_DIM + col] = acc[nc][r] + bb;
    }
  }
}

// ---------------------------------------------------------------------------
// Grid barrier for the persistent scan kernel (monotonic counter in d_ws)
// ---------------------------------------------------------------------------
__device__ __forceinline__ void grid_barrier(unsigned* cnt, unsigned target) {
  __syncthreads();
  if (threadIdx.x == 0) {
    __hip_atomic_fetch_add(cnt, 1u, __ATOMIC_RELEASE, __HIP_MEMORY_SCOPE_AGENT);
    while (__hip_atomic_load(cnt, __ATOMIC_ACQUIRE, __HIP_MEMORY_SCOPE_AGENT) <
           target) {
      __builtin_amdgcn_s_sleep(2);
    }
  }
  __syncthreads();
  __threadfence();
}

// ---------------------------------------------------------------------------
// Recurrence: for t in 0..T-1:
//   h = tanh(xin_t + h @ W1h); out_t = h @ W2 (+b2 folded in after the max)
//   pooled = max_t out_t
// 8 persistent WGs, each owns 64 columns of H. W1h/W2 slices + full h live in
// LDS (~195 KB of the 320 KB/WGP). h is published per step as bf16 (64 KB,
// L2-resident); the re-broadcast into LDS uses the Tensor Data Mover with
// hardware row padding (512 elems -> KP=520), one DMA per step per WG.
// ---------------------------------------------------------------------------
__global__ __launch_bounds__(256) void rnn_scan(
    const float* __restrict__ xin, const float* __restrict__ W1,
    const float* __restrict__ W2, const float* __restrict__ b2,
    void* __restrict__ hbuf_v, float* __restrict__ pooled,
    unsigned* __restrict__ bar) {
  extern __shared__ char smem_raw[];
  __bf16* sWh = (__bf16*)smem_raw;       // [64 n][KP k]  (transposed W1h slice)
  __bf16* sW2 = sWh + 64 * KP;           // [64 n][KP k]  (transposed W2 slice)
  __bf16* sH  = sW2 + 64 * KP;           // [64 b][KP k]  (current h, bf16)
  __bf16* hbuf = (__bf16*)hbuf_v;        // global 64x512 bf16 h exchange buffer

  const int tid  = threadIdx.x;
  const int lane = tid & 31;
  const int wave = tid >> 5;
  const int c0   = blockIdx.x * 64;      // owned H columns [c0, c0+64)

  // Preload transposed weight slices into LDS (once), convert f32 -> bf16
  for (int e = tid; e < 64 * H_DIM; e += 256) {
    const int n = e >> 9, k = e & 511;
    sWh[n * KP + k] = (__bf16)W1[(E_DIM + k) * H_DIM + c0 + n];
    sW2[n * KP + k] = (__bf16)W2[k * H_DIM + c0 + n];
  }
  for (int e = tid; e < 64 * KP; e += 256) sH[e] = (__bf16)0.0f;  // h0 = 0
  __syncthreads();

#if USE_TDM
  // Tensor DMA descriptor (ISA 8.3/8.4): hbuf (64x512 bf16, stride 512) ->
  // sH with 4-DWORD padding after every 256 DWORDs (row 1024B -> 1040B = KP).
  const unsigned long long ga = (unsigned long long)(uintptr_t)hbuf;
  const unsigned lds_sh =
      __builtin_amdgcn_groupstaticsize() + 2u * 64u * KP * sizeof(__bf16);
  v4u g0;
  g0[0] = 1u;                                   // count=1, user descriptor
  g0[1] = lds_sh;                               // lds_addr (bytes)
  g0[2] = (unsigned)ga;                         // global_addr[31:0]
  g0[3] = (unsigned)(ga >> 32) | 0x80000000u;   // global_addr[56:32] | type=2
  v8i_t g1;
  g1[0] = (int)((1u << 16)      // data_size = 2 bytes
              | (1u << 20)      // pad_enable
              | (7u << 22)      // pad_interval: every 256 DWORDs
              | (3u << 25));    // pad_amount: 4 DWORDs (16 B)
  g1[1] = (int)(512u << 16);    // tensor_dim0 = 512 (bits 63:48)
  g1[2] = (int)(64u << 16);     // tensor_dim1 = 64  (bits 95:80)
  g1[3] = (int)(512u << 16);    // tile_dim0 = 512   (bits 127:112)
  g1[4] = 64;                   // tile_dim1 = 64    (bits 143:128)
  g1[5] = 512;                  // tensor_dim0_stride = 512 (bits 207:160)
  g1[6] = 0;
  g1[7] = 0;
  v4i_t gz4;
  gz4[0] = 0; gz4[1] = 0; gz4[2] = 0; gz4[3] = 0;
  v8i_t gz8;
#pragma unroll
  for (int i = 0; i < 8; ++i) gz8[i] = 0;
#endif

  // Wave -> two 16x16 tiles of the 64x64 output slice
  const int mr   = wave >> 1;            // row tile 0..3
  const int nc0  = (wave & 1) * 2;       // col tiles nc0, nc0+1
  const int m0   = mr * 16;
  const int mrow = m0 + ((lane & 16) ? 8 : 0);
  const int ncol = lane & 15;
  const int cA   = c0 + nc0 * 16 + ncol;
  const int cB   = c0 + (nc0 + 1) * 16 + ncol;

  v8f mx0 = vsplat8(-3.402823466e38f);
  v8f mx1 = vsplat8(-3.402823466e38f);
  unsigned epoch = 0;

  for (int t = 0; t < T_DIM; ++t) {
    // z = h @ W1h (our 64x64 slice)
    v8f z0 = vzero8(), z1 = vzero8();
    for (int k0 = 0; k0 < H_DIM; k0 += 32) {
      const v16bf a  = frag_ld(&sH[m0 * KP + k0], KP, lane);
      const v16bf b0 = frag_ld(&sWh[(nc0 * 16) * KP + k0], KP, lane);
      const v16bf b1 = frag_ld(&sWh[((nc0 + 1) * 16) * KP + k0], KP, lane);
      z0 = wmma_bf16(a, b0, z0);
      z1 = wmma_bf16(a, b1, z1);
    }
    // h_new = tanh(xin_t + z), publish as bf16
    const float* xt = xin + (size_t)t * B_DIM * H_DIM;
#pragma unroll
    for (int r = 0; r < 8; ++r) {
      const int row = mrow + r;
      const float h0v = tanhf(z0[r] + xt[row * H_DIM + cA]);
      const float h1v = tanhf(z1[r] + xt[row * H_DIM + cB]);
      hbuf[row * H_DIM + cA] = (__bf16)h0v;
      hbuf[row * H_DIM + cB] = (__bf16)h1v;
    }
    __threadfence();
    epoch += NWG_REC;
    grid_barrier(bar, epoch);            // all slices of h_new published

    // Refresh full h into LDS
#if USE_TDM
    if (tid == 0) {
      __builtin_amdgcn_tensor_load_to_lds(g0, g1, gz4, gz4, gz8, 0);
      __builtin_amdgcn_s_wait_tensorcnt((short)0);
    }
#else
    for (int e = tid; e < B_DIM * H_DIM; e += 256) {
      const int brow = e >> 9, k = e & 511;
      sH[brow * KP + k] = hbuf[e];
    }
#endif
    epoch += NWG_REC;
    grid_barrier(bar, epoch);            // LDS copy done + all WGs past hbuf

    // out_t slice = h_new @ W2; fold into running max
    v8f o0 = vzero8(), o1 = vzero8();
    for (int k0 = 0; k0 < H_DIM; k0 += 32) {
      const v16bf a  = frag_ld(&sH[m0 * KP + k0], KP, lane);
      const v16bf b0 = frag_ld(&sW2[(nc0 * 16) * KP + k0], KP, lane);
      const v16bf b1 = frag_ld(&sW2[((nc0 + 1) * 16) * KP + k0], KP, lane);
      o0 = wmma_bf16(a, b0, o0);
      o1 = wmma_bf16(a, b1, o1);
    }
#pragma unroll
    for (int r = 0; r < 8; ++r) {
      mx0[r] = fmaxf(mx0[r], o0[r]);
      mx1[r] = fmaxf(mx1[r], o1[r]);
    }
  }

  // pooled = max_t out_t + b2   (b2 is t-invariant: add once after the max)
#pragma unroll
  for (int r = 0; r < 8; ++r) {
    const int row = mrow + r;
    pooled[row * H_DIM + cA] = mx0[r] + b2[cA];
    pooled[row * H_DIM + cB] = mx1[r] + b2[cB];
  }
}

// ---------------------------------------------------------------------------
// Head: out[b][c] = pooled[b] . Wc[:,c] + bc[c]   (64x512x8, tiny)
// ---------------------------------------------------------------------------
__global__ __launch_bounds__(512) void head_gemm(
    const float* __restrict__ pooled, const float* __restrict__ Wc,
    const float* __restrict__ bc, float* __restrict__ out) {
  const int tid = threadIdx.x;
  const int b = tid >> 3, c = tid & 7;
  float s = bc[c];
#pragma unroll 8
  for (int h = 0; h < H_DIM; ++h) s += pooled[b * H_DIM + h] * Wc[h * C_DIM + c];
  out[b * C_DIM + c] = s;
}

// ---------------------------------------------------------------------------
// Launch
// ---------------------------------------------------------------------------
extern "C" void kernel_launch(void* const* d_in, const int* in_sizes, int n_in,
                              void* d_out, int out_size, void* d_ws,
                              size_t ws_size, hipStream_t stream) {
  (void)in_sizes; (void)n_in; (void)out_size; (void)ws_size;
  const int*   x   = (const int*)d_in[0];
  const float* emb = (const float*)d_in[1];
  const float* W1  = (const float*)d_in[2];
  const float* b1  = (const float*)d_in[3];
  const float* W2  = (const float*)d_in[4];
  const float* b2  = (const float*)d_in[5];
  const float* Wc  = (const float*)d_in[6];
  const float* bc  = (const float*)d_in[7];
  float* out = (float*)d_out;

  // Workspace layout
  char* ws = (char*)d_ws;
  unsigned* bar    = (unsigned*)ws;                        //    256 B (counter)
  void*     hbuf   = (void*)(ws + 256);                    // 65,536 B (bf16 h)
  float*    pooled = (float*)(ws + 256 + 65536);           // 131,072 B
  float*    xin    = (float*)(ws + 256 + 65536 + 131072);  // 64 MB (T,B,H) f32

  // Reset barrier counter and h0 = 0 every launch (graph-capture safe)
  (void)hipMemsetAsync(ws, 0, 256 + 65536, stream);

  // Phase 1: gather + (32768 x 512 x 512) GEMM
  phase1_gemm<<<dim3((B_DIM * T_DIM) / 64, H_DIM / 64), 128, 0, stream>>>(
      x, emb, W1, b1, xin);

  // Phase 2: persistent cooperative scan (8 WGs, ~195 KB dynamic LDS each)
  const size_t lds_bytes = (size_t)3 * 64 * KP * sizeof(__bf16);
  rnn_scan<<<NWG_REC, 256, lds_bytes, stream>>>(xin, W1, W2, b2, hbuf, pooled,
                                                bar);

  // Phase 3: head
  head_gemm<<<1, 512, 0, stream>>>(pooled, Wc, bc, out);
}
